// UNetSCN_ED_40948218200124
// MI455X (gfx1250) — matrix operand order repaired and checked by reading, not compiled, not run on hardware
//
#include <hip/hip_runtime.h>
#include <hip/hip_bf16.h>

// ============================================================================
// Masked 3-D U-Net (scn-style) for gfx1250 / MI455X — round 4.
//
// Conv = implicit GEMM on v_wmma_f32_16x16x32_bf16, wave32, one wave/block,
// 16(M) x 32(N) output tile per wave (two accumulators / two WMMAs per K pair,
// K-pair loop unrolled x2 -> 4 WMMAs in flight to hide WMMA->WMMA RAW).
//  * K = taps x Cin in 16-wide chunks; tap/chunk split incremental, site
//    decode shift/mask (grids are powers of two) -> no integer division.
//  * Weights pre-packed per layer into bf16 B-operand layout
//    ([chunk][n][16 K contiguous]) -> one 16B load per lane per chunk (its K half).
//  * A loads are 2x float4; 3-channel stem takes a scalar fallback.
//  * x pre-masked once; output mask fused in the store epilogue.
// BN stats: LDS partial sums (ds_add_f32) + one global atomic per channel per
// block; BN+ReLU+mask fused elementwise. Skip-concat copy routes each 16B
// chunk through the gfx1250 async LDS data mover (ASYNCcnt + s_wait_asynccnt;
// builtin signature confirmed by round-3 diagnostics: (v4i AS1*, v4i AS3*,
// imm offset, imm cpol)), with a float4 fallback if the builtins are absent.
//
// Input pointer layout assumption (params pytree flattened, sorted dict keys):
//   0: x (1,64,64,64,3) f32        1: mask (1,64,64,64) bool (1 byte)
//   2..4   dec1:  b,g,ws
//   5..34  dec2..dec6: b1,b2,g1,g2,ws,wt   (6 each, base = 5+(idx-2)*6)
//   35..37 dec7:  b,g,wt
//   38..73 enc2..enc7: b1,b2,g1,g2,wd,ws   (6 each, base = 38+(i-2)*6)
//   74..76 mb1:   b,g,w
//   77..78 out:   b,g
//   79     w_in (3,3,3,3,16)
// ============================================================================

typedef __attribute__((ext_vector_type(16))) __bf16 v16bf;
typedef __attribute__((ext_vector_type(8)))  float  v8f;
typedef __attribute__((ext_vector_type(4)))  int    v4i;

#define BN_EPS 1e-4f

#if __has_builtin(__builtin_amdgcn_global_load_async_to_lds_b128) &&   \
    __has_builtin(__builtin_amdgcn_global_store_async_from_lds_b128) && \
    __has_builtin(__builtin_amdgcn_s_wait_asynccnt)
#define USE_ASYNC_LDS 1
#else
#define USE_ASYNC_LDS 0
#endif

// ---------------------------------------------------------------------------
// mask -> float occupancy
__global__ void mask_to_float_kernel(const unsigned char* __restrict__ msk,
                                     int n, float* __restrict__ mf) {
  for (int i = blockIdx.x * blockDim.x + threadIdx.x; i < n;
       i += gridDim.x * blockDim.x)
    mf[i] = msk[i] ? 1.0f : 0.0f;
}

// 2x2x2 "any" pooling of the occupancy mask (Nout = 1<<ns)
__global__ void pool_mask_kernel(const float* __restrict__ in, int ns,
                                 float* __restrict__ out) {
  const int Nout = 1 << ns;
  const int nsIn = ns + 1;
  const int Sout = 1 << (3 * ns);
  const int mask = Nout - 1;
  for (int i = blockIdx.x * blockDim.x + threadIdx.x; i < Sout;
       i += gridDim.x * blockDim.x) {
    const int w = i & mask;
    const int h = (i >> ns) & mask;
    const int d = i >> (2 * ns);
    float s = 0.f;
    for (int dz = 0; dz < 2; ++dz)
      for (int dy = 0; dy < 2; ++dy)
        for (int dx = 0; dx < 2; ++dx)
          s += in[((((2 * d + dz) << nsIn) + (2 * h + dy)) << nsIn) +
                  (2 * w + dx)];
    out[i] = (s > 0.f) ? 1.0f : 0.0f;
  }
}

// count of active sites (LDS tree reduce + one global atomic per block)
__global__ void count_mask_kernel(const float* __restrict__ mf, int n,
                                  float* __restrict__ cnt) {
  __shared__ float sm[256];
  float v = 0.f;
  for (int i = blockIdx.x * blockDim.x + threadIdx.x; i < n;
       i += gridDim.x * blockDim.x)
    v += mf[i];
  sm[threadIdx.x] = v;
  __syncthreads();
  for (int s = 128; s > 0; s >>= 1) {
    if ((int)threadIdx.x < s) sm[threadIdx.x] += sm[threadIdx.x + s];
    __syncthreads();
  }
  if (threadIdx.x == 0) atomicAdd(cnt, sm[0]);
}

// pre-mask the stem input: y[site,c] = x[site,c] * mf[site]
__global__ void premask_kernel(const float* __restrict__ x,
                               const float* __restrict__ mf, int S, int C,
                               float* __restrict__ y) {
  for (int site = blockIdx.x * blockDim.x + threadIdx.x; site < S;
       site += gridDim.x * blockDim.x) {
    const float mv = mf[site];
    const int base = site * C;
    for (int c = 0; c < C; ++c) y[base + c] = x[base + c] * mv;
  }
}

// ---------------------------------------------------------------------------
// BN statistics: per-channel masked sum and sum-of-squares (site-major).
__global__ void bn_sum_kernel(const float* __restrict__ x,
                              const float* __restrict__ mf, int S, int C,
                              float* __restrict__ sums) {
  extern __shared__ float sm[];  // 2*C floats
  for (int i = threadIdx.x; i < 2 * C; i += blockDim.x) sm[i] = 0.f;
  __syncthreads();
  for (int site = blockIdx.x * blockDim.x + threadIdx.x; site < S;
       site += gridDim.x * blockDim.x) {
    if (mf[site] == 0.f) continue;  // inactive sites contribute nothing
    const int base = site * C;
    for (int c = 0; c < C; ++c) {
      const float v = x[base + c];
      atomicAdd(&sm[c], v);
      atomicAdd(&sm[C + c], v * v);
    }
  }
  __syncthreads();
  for (int i = threadIdx.x; i < 2 * C; i += blockDim.x)
    if (sm[i] != 0.f) atomicAdd(&sums[i], sm[i]);
}

// fused BN + ReLU + re-mask:  y = relu((x-mean)*g*rsqrt(var+eps)+b) * mf
__global__ void bn_apply_kernel(const float* __restrict__ x,
                                const float* __restrict__ mf, int S, int C,
                                const float* __restrict__ sums,
                                const float* __restrict__ cnt,
                                const float* __restrict__ g,
                                const float* __restrict__ bb,
                                float* __restrict__ y) {
  const float n = fmaxf(cnt[0], 1.0f);
  for (int site = blockIdx.x * blockDim.x + threadIdx.x; site < S;
       site += gridDim.x * blockDim.x) {
    const float mv = mf[site];
    const int base = site * C;
    for (int c = 0; c < C; ++c) {
      const float mean = sums[c] / n;
      const float var = sums[C + c] / n - mean * mean;
      const float v =
          (x[base + c] - mean) * (g[c] * rsqrtf(var + BN_EPS)) + bb[c];
      y[base + c] = fmaxf(v, 0.0f) * mv;
    }
  }
}

// ---------------------------------------------------------------------------
// Skip-concat copy, 16B granularity (Ca, Ct, off are multiples of 16 floats
// -> 64B aligned). Each chunk goes through the gfx1250 async LDS data mover
// (GLOBAL_LOAD_ASYNC_TO_LDS_B128 / GLOBAL_STORE_ASYNC_FROM_LDS_B128, ASYNCcnt).
__global__ void copy_cat_kernel(const float* __restrict__ src, int S, int Ca,
                                float* __restrict__ dst, int Ct, int off) {
  const int q = Ca >> 2;  // 16B chunks per site
#if USE_ASYNC_LDS
  __shared__ v4i stage[256];
#endif
  for (int site = blockIdx.x * blockDim.x + threadIdx.x; site < S;
       site += gridDim.x * blockDim.x) {
    const float4* sp = (const float4*)(src + site * Ca);
    float4* dp = (float4*)(dst + site * Ct + off);
    for (int c = 0; c < q; ++c) {
#if USE_ASYNC_LDS
      __builtin_amdgcn_global_load_async_to_lds_b128(
          (__attribute__((address_space(1))) v4i*)(sp + c),
          (__attribute__((address_space(3))) v4i*)&stage[threadIdx.x], 0, 0);
      __builtin_amdgcn_s_wait_asynccnt(0);
      __builtin_amdgcn_global_store_async_from_lds_b128(
          (__attribute__((address_space(1))) v4i*)(dp + c),
          (__attribute__((address_space(3))) v4i*)&stage[threadIdx.x], 0, 0);
      __builtin_amdgcn_s_wait_asynccnt(0);
#else
      dp[c] = sp[c];
#endif
    }
  }
}

// ---------------------------------------------------------------------------
// Pack conv weights (DHWIO f32) into the WMMA B-operand layout:
//   bp[((t*cpt + k)*Cout + n)*16 + e] = bf16( W[t, k*16+e, n] )  (0 if >= Cin)
__global__ void pack_w_kernel(const float* __restrict__ w, int nt, int Cin,
                              int Cout, __bf16* __restrict__ bp) {
  const int cpt = (Cin + 15) >> 4;
  const int rows = nt * cpt * Cout;
  for (int r = blockIdx.x * blockDim.x + threadIdx.x; r < rows;
       r += gridDim.x * blockDim.x) {
    const int n = r % Cout;  // tiny kernel; division acceptable here
    const int tk = r / Cout;
    const int k = tk % cpt;
    const int t = tk / cpt;
    const int out = r * 16;
    for (int e = 0; e < 16; ++e) {
      const int ch = k * 16 + e;
      bp[out + e] =
          (ch < Cin) ? (__bf16)w[(t * Cin + ch) * Cout + n] : (__bf16)0.0f;
    }
  }
}

// ---------------------------------------------------------------------------
// Generic masked conv as implicit GEMM on WMMA (one wave per block).
//   TYPE 0: 3^3 submanifold (SAME, 27 taps, spatial bounds check)
//   TYPE 1: 2^3 stride-2 down (VALID, 8 taps, never OOB)
//   TYPE 2: 2^3 stride-2 deconv (8 taps, only parity-matching tap live)
// 16x32 output tile per wave: acc0 (cols nCol0..+15), acc1 (nCol1..+15).
template <int TYPE>
__global__ __launch_bounds__(32)
void conv_wmma_kernel(const float* __restrict__ in,
                      const __bf16* __restrict__ bp,
                      const float* __restrict__ mfout, float* __restrict__ out,
                      int ns /*log2 Nout*/, int Cin, int Cout, int outStride,
                      int chanOff) {
  const int Nout = 1 << ns;
  const int Sout = 1 << (3 * ns);
  const int lane = threadIdx.x;  // wave32
  const int mlane = lane & 15;
  const int half = lane >> 4;
  const int tileM = blockIdx.x;
  const int tileN = blockIdx.y;

  const int nt = (TYPE == 0) ? 27 : 8;
  const int cpt = (Cin + 15) >> 4;  // 16-wide K chunks per tap
  const int totalChunks = nt * cpt;
  const int npairs = (totalChunks + 1) >> 1;
  const bool cinAligned = (Cin & 15) == 0;

  // A operand: this lane serves output row (tileM*16 + mlane).
  const int rowA = tileM * 16 + mlane;
  const bool rowOk = rowA < Sout;
  const int dmask = Nout - 1;
  const int ow = rowA & dmask;
  const int oh = (rowA >> ns) & dmask;
  const int od = (rowA >> (2 * ns)) & dmask;
  const int parity = ((od & 1) << 2) | ((oh & 1) << 1) | (ow & 1);

  const int nsIn = (TYPE == 0) ? ns : ((TYPE == 1) ? ns + 1 : ns - 1);
  const int Nin = 1 << nsIn;

  const int nCol0 = tileN * 32 + mlane;  // B / D columns
  const int nCol1 = nCol0 + 16;
  const bool hasN1 = nCol1 < Cout;       // uniform across the wave

  v8f acc0 = {}, acc1 = {};
  int t = 0, k = 0;  // incremental tap / chunk-in-tap counters
#pragma unroll 2
  for (int jp = 0; jp < npairs; ++jp) {
    v16bf a;
    v16bf b0 = {}, b1 = {};
#pragma unroll
    for (int piece = 0; piece < 2; ++piece) {
      const int c = 2 * jp + piece;
      if (c < totalChunks) {  // uniform
        // ---- neighbour coords for tap t
        int id = 0, ih = 0, iw = 0;
        bool valid = rowOk;
        if (TYPE == 0) {
          id = od + (t / 9) - 1;
          ih = oh + ((t / 3) % 3) - 1;  // /9,/3 are constant divisions
          iw = ow + (t % 3) - 1;
          valid = valid && (unsigned)id < (unsigned)Nin &&
                  (unsigned)ih < (unsigned)Nin && (unsigned)iw < (unsigned)Nin;
        } else if (TYPE == 1) {
          id = 2 * od + (t >> 2);
          ih = 2 * oh + ((t >> 1) & 1);
          iw = 2 * ow + (t & 1);
        } else {  // deconv: exactly one tap (output parity) is live
          id = od >> 1;
          ih = oh >> 1;
          iw = ow >> 1;
          valid = valid && (t == parity);
        }
        // ---- A: 8 channels of the neighbour site (this lane's K half)
        if (valid) {
          const int site = (((id << nsIn) + ih) << nsIn) + iw;
          const int chA = (k << 4) + (half << 3);
          const float* src = in + site * Cin + chA;
          if (cinAligned) {  // uniform: 2x float4 (32B-aligned)
            const float4 v0 = *(const float4*)src;
            const float4 v1 = *(const float4*)(src + 4);
            a[piece * 8 + 0] = (__bf16)v0.x;
            a[piece * 8 + 1] = (__bf16)v0.y;
            a[piece * 8 + 2] = (__bf16)v0.z;
            a[piece * 8 + 3] = (__bf16)v0.w;
            a[piece * 8 + 4] = (__bf16)v1.x;
            a[piece * 8 + 5] = (__bf16)v1.y;
            a[piece * 8 + 6] = (__bf16)v1.z;
            a[piece * 8 + 7] = (__bf16)v1.w;
          } else {  // stem (Cin=3)
#pragma unroll
            for (int e = 0; e < 8; ++e)
              a[piece * 8 + e] =
                  (chA + e < Cin) ? (__bf16)src[e] : (__bf16)0.0f;
          }
        } else {
#pragma unroll
          for (int e = 0; e < 8; ++e) a[piece * 8 + e] = (__bf16)0.0f;
        }
        // ---- B: this lane's 8 packed K rows of this chunk (K half = half)
        {
          const __bf16* pb0 = bp + (c * Cout + nCol0) * 16 + (half << 3);
          const __bf16* pb1 = bp + (c * Cout + nCol1) * 16 + (half << 3);
#pragma unroll
          for (int e = 0; e < 8; ++e) {
            b0[piece * 8 + e] = pb0[e];
            if (hasN1) b1[piece * 8 + e] = pb1[e];
          }
          if (c + 2 < totalChunks)  // gfx1250 global_prefetch_b8
            __builtin_prefetch((const void*)(bp + ((c + 2) * Cout + nCol0) * 16),
                               0, 3);
        }
        if (++k == cpt) { k = 0; ++t; }
      } else {  // padded final chunk: zero A half; B half stays zero
#pragma unroll
        for (int e = 0; e < 8; ++e) a[piece * 8 + e] = (__bf16)0.0f;
      }
    }
    acc0 = __builtin_amdgcn_wmma_f32_16x16x32_bf16(
        false, a, false, b0, (short)0, acc0, false, false);
    if (hasN1)
      acc1 = __builtin_amdgcn_wmma_f32_16x16x32_bf16(
          false, a, false, b1, (short)0, acc1, false, false);
  }

  // D layout: lane = column, VGPR r = row r + 8*half. Fuse output mask.
#pragma unroll
  for (int r = 0; r < 8; ++r) {
    const int site = tileM * 16 + r + (half << 3);
    if (site < Sout) {
      const float mv = mfout[site];
      const int base = site * outStride + chanOff;
      out[base + nCol0] = acc0[r] * mv;
      if (hasN1) out[base + nCol1] = acc1[r] * mv;
    }
  }
}

// ===========================================================================
extern "C" void kernel_launch(void* const* d_in, const int* in_sizes, int n_in,
                              void* d_out, int out_size, void* d_ws,
                              size_t ws_size, hipStream_t stream) {
  (void)in_sizes; (void)n_in; (void)out_size; (void)ws_size;

  const float* x = (const float*)d_in[0];
  const unsigned char* mask = (const unsigned char*)d_in[1];
  auto P = [&](int i) { return (const float*)d_in[i]; };

  // Param index map (see header comment).
  const int DEC1_B = 2, DEC1_G = 3, DEC1_WS = 4;
  const int DEC7_B = 35, DEC7_G = 36, DEC7_WT = 37;
  const int MB1_B = 74, MB1_G = 75, MB1_W = 76;
  const int OUT_B = 77, OUT_G = 78, W_IN = 79;
  auto DEC = [&](int idx) { return 5 + (idx - 2) * 6; };  // b1,b2,g1,g2,ws,wt
  auto ENC = [&](int i) { return 38 + (i - 2) * 6; };     // b1,b2,g1,g2,wd,ws

  int dims[7], ns[7];
  long S[7];
  for (int l = 0; l < 7; ++l) {
    ns[l] = 6 - l;
    dims[l] = 1 << ns[l];
    S[l] = (long)dims[l] * dims[l] * dims[l];
  }

  // ---- workspace layout (floats, 256B-aligned chunks) ----
  float* W = (float*)d_ws;
  size_t cur = 0;
  auto alloc = [&](long n) {
    float* p = W + cur;
    cur += ((size_t)n + 63) & ~(size_t)63;
    return p;
  };
  float* mf[7];
  for (int l = 0; l < 7; ++l) mf[l] = alloc(S[l]);
  float* cnt = alloc(8);           // active-site count per level
  float* bns = alloc(512);         // BN scratch: [sum(0..C), sumsq(C..2C)]
  float* xm = alloc(S[0] * 3);     // pre-masked stem input
  __bf16* wpack = (__bf16*)alloc(256000);  // packed weights (max ~498K bf16)
  float* f[8];
  for (int i = 1; i <= 7; ++i) f[i] = alloc(S[i - 1] * (i * 16));
  float* tA = alloc(S[0] * 32);
  float* tB = alloc(S[0] * 32);
  float* cat[6];
  cat[5] = alloc(S[5] * 192); cat[4] = alloc(S[4] * 160);
  cat[3] = alloc(S[3] * 128); cat[2] = alloc(S[2] * 96);
  cat[1] = alloc(S[1] * 64);  cat[0] = alloc(S[0] * 32);

  auto EW = [](long n) {
    long b = (n + 255) / 256;
    if (b > 4096) b = 4096;
    if (b < 1) b = 1;
    return (unsigned)b;
  };

  // ---- occupancy masks per level + active-site counts ----
  mask_to_float_kernel<<<EW(S[0]), 256, 0, stream>>>(mask, (int)S[0], mf[0]);
  for (int l = 1; l < 7; ++l)
    pool_mask_kernel<<<EW(S[l]), 256, 0, stream>>>(mf[l - 1], ns[l], mf[l]);
  (void)hipMemsetAsync(cnt, 0, 8 * sizeof(float), stream);
  for (int l = 0; l < 7; ++l)
    count_mask_kernel<<<EW(S[l]), 256, 0, stream>>>(mf[l], (int)S[l], cnt + l);
  premask_kernel<<<EW(S[0]), 256, 0, stream>>>(x, mf[0], (int)S[0], 3, xm);

  // ---- helpers ----
  auto bn = [&](const float* xin, int lv, int C, const float* g,
                const float* bb, float* yout) {
    (void)hipMemsetAsync(bns, 0, 2 * C * sizeof(float), stream);
    bn_sum_kernel<<<EW(S[lv]), 256, 2 * C * sizeof(float), stream>>>(
        xin, mf[lv], (int)S[lv], C, bns);
    bn_apply_kernel<<<EW(S[lv]), 256, 0, stream>>>(xin, mf[lv], (int)S[lv], C,
                                                   bns, cnt + lv, g, bb, yout);
  };
  auto conv = [&](int type, const float* xin, const float* wgt, int lvout,
                  int Cin, int Cout, float* outp, int outStride, int chanOff) {
    const int nt = (type == 0) ? 27 : 8;
    const int cpt = (Cin + 15) >> 4;
    pack_w_kernel<<<EW((long)nt * cpt * Cout), 256, 0, stream>>>(wgt, nt, Cin,
                                                                 Cout, wpack);
    dim3 grid((unsigned)((S[lvout] + 15) / 16), (unsigned)((Cout + 31) / 32));
    switch (type) {
      case 0:
        conv_wmma_kernel<0><<<grid, 32, 0, stream>>>(
            xin, wpack, mf[lvout], outp, ns[lvout], Cin, Cout, outStride,
            chanOff);
        break;
      case 1:
        conv_wmma_kernel<1><<<grid, 32, 0, stream>>>(
            xin, wpack, mf[lvout], outp, ns[lvout], Cin, Cout, outStride,
            chanOff);
        break;
      default:
        conv_wmma_kernel<2><<<grid, 32, 0, stream>>>(
            xin, wpack, mf[lvout], outp, ns[lvout], Cin, Cout, outStride,
            chanOff);
        break;
    }
  };

  // ---- stem: h = conv3(x*mf0, w_in)*mf0 ; f1 = conv3(bnrelu(h), mb1.w)*mf0
  conv(0, xm, P(W_IN), 0, 3, 16, tA, 16, 0);
  bn(tA, 0, 16, P(MB1_G), P(MB1_B), tB);
  conv(0, tB, P(MB1_W), 0, 16, 16, f[1], 16, 0);

  // ---- encoder ----
  for (int i = 2; i <= 7; ++i) {
    const int lv = i - 1;
    const int a = (i - 1) * 16, c = i * 16;
    const int bse = ENC(i);
    bn(f[i - 1], lv - 1, a, P(bse + 2), P(bse + 0), tA);  // g1,b1
    conv(1, tA, P(bse + 4), lv, a, c, tB, c, 0);          // wd (down)
    bn(tB, lv, c, P(bse + 3), P(bse + 1), tA);            // g2,b2
    conv(0, tA, P(bse + 5), lv, c, c, f[i], c, 0);        // ws (subm)
  }

  // ---- dec7: deconv 7m->6m into cat5[96..191], skip f6 into cat5[0..95]
  bn(f[7], 6, 112, P(DEC7_G), P(DEC7_B), tA);
  conv(2, tA, P(DEC7_WT), 5, 112, 96, cat[5], 192, 96);
  copy_cat_kernel<<<EW(S[5]), 256, 0, stream>>>(f[6], (int)S[5], 96, cat[5],
                                                192, 0);

  // ---- dec6..dec2 ----
  const int decIdx[5] = {6, 5, 4, 3, 2};
  const int decLv[5] = {5, 4, 3, 2, 1};
  for (int s = 0; s < 5; ++s) {
    const int idx = decIdx[s], lv = decLv[s];
    const int a = (lv + 1) * 32;  // 192,160,128,96,64 = cat[lv] channels
    const int mid = a / 2;
    const int cf = lv * 16;       // skip-feature channels at level lv-1
    const int bse = DEC(idx);
    bn(cat[lv], lv, a, P(bse + 2), P(bse + 0), tA);            // g1,b1
    conv(0, tA, P(bse + 4), lv, a, mid, tB, mid, 0);           // ws (subm)
    bn(tB, lv, mid, P(bse + 3), P(bse + 1), tA);               // g2,b2
    conv(2, tA, P(bse + 5), lv - 1, mid, cf, cat[lv - 1], 2 * cf, cf);
    copy_cat_kernel<<<EW(S[lv - 1]), 256, 0, stream>>>(
        f[lv], (int)S[lv - 1], cf, cat[lv - 1], 2 * cf, 0);
  }

  // ---- dec1 + output BN/ReLU ----
  bn(cat[0], 0, 32, P(DEC1_G), P(DEC1_B), tA);
  conv(0, tA, P(DEC1_WS), 0, 32, 16, tB, 16, 0);
  bn(tB, 0, 16, P(OUT_G), P(OUT_B), (float*)d_out);
}
